// HyperNetwork_FC_APG_Fusion_VQVAE_68874095558713
// MI455X (gfx1250) — matrix operand order repaired and checked by reading, not compiled, not run on hardware
//
#include <hip/hip_runtime.h>
#include <hip/hip_bf16.h>

typedef __attribute__((ext_vector_type(16))) _Float16 v16h;
typedef __attribute__((ext_vector_type(8)))  float    v8f;

#define CB   256      // batch
#define CS   50       // seq
#define CD   32       // latent dim
#define CV   10000    // codes
#define CVP  10240    // padded codes (multiple of 128)
#define CK   16       // hyper K
#define CH   64       // decoder hidden
#define CL   2        // hyper layers
#define CU   32       // in/out dim
#define CTRIG 30
#define CNORM 0.01f
#define CBS  (CB*CS)          // 12800 rows
#define MTILE (CBS/16)        // 800
#define CHUNK 128             // codes per LDS chunk (8KB f16)
#define NCHUNK (CVP/CHUNK)    // 80

__device__ __forceinline__ float dot32(const float* __restrict__ a,
                                       const float* __restrict__ b) {
  float s = 0.f;
#pragma unroll
  for (int i = 0; i < 32; ++i) s += a[i] * b[i];
  return s;
}
__device__ __forceinline__ float sigm(float x) { return 1.0f / (1.0f + __expf(-x)); }

// ---- CDNA5 async copy: 16B global -> LDS, tracked by ASYNCcnt ---------------
__device__ __forceinline__ void async_cp16(unsigned int lds_byte, const void* g) {
  asm volatile("global_load_async_to_lds_b128 %0, %1, off"
               :: "v"(lds_byte), "v"((unsigned long long)(uintptr_t)g)
               : "memory");
}
__device__ __forceinline__ void wait_async0() {
  asm volatile("s_wait_asynccnt 0x0" ::: "memory");
}

// ---------------- K0: convert embd to f16 (padded), half-norms ---------------
__global__ void k_pack(const float* __restrict__ embd, _Float16* __restrict__ embd_h,
                       float* __restrict__ hn) {
  int v = blockIdx.x * blockDim.x + threadIdx.x;
  if (v >= CVP) return;
  if (v < CV) {
    float s = 0.f;
#pragma unroll
    for (int d = 0; d < CD; ++d) {
      float e = embd[v * CD + d];
      embd_h[v * CD + d] = (_Float16)e;
      s += e * e;
    }
    hn[v] = 0.5f * s;
  } else {
#pragma unroll
    for (int d = 0; d < CD; ++d) embd_h[v * CD + d] = (_Float16)0.f;
    hn[v] = 1.0e30f;   // padded codes can never win the argmax
  }
}

// ---------------- K1: VQ argmax, WMMA f16 + async LDS double buffer ----------
// block = 256 thr = 8 waves; wave w owns M-tile blockIdx*8+w; codes streamed
// through LDS in 128-code chunks via global_load_async_to_lds_b128.
__global__ void __launch_bounds__(256)
k_vq(const float* __restrict__ z1, const _Float16* __restrict__ embd_h,
     const float* __restrict__ hn, int* __restrict__ idx_out) {
  __shared__ __align__(32) _Float16 ebuf[2][CHUNK * CD];  // 2 x 8KB
  __shared__ __align__(16) float    hbuf[2][CHUNK];       // 2 x 512B

  const int tid  = threadIdx.x;
  const int wave = tid >> 5;
  const int lane = tid & 31;
  const int n    = lane & 15;
  const int kh   = lane >> 4;
  const int m0   = (blockIdx.x * 8 + wave) * 16;

  const unsigned int eb0 = (unsigned int)(uintptr_t)&ebuf[0][0];
  const unsigned int eb1 = (unsigned int)(uintptr_t)&ebuf[1][0];
  const unsigned int hb0 = (unsigned int)(uintptr_t)&hbuf[0][0];
  const unsigned int hb1 = (unsigned int)(uintptr_t)&hbuf[1][0];

  // Pack A (16x32 f16): lane holds row m=n; low lanes K in [0,16),
  // high lanes K in [16,32)  (contiguous, matches B)
  const float* qrow = z1 + (size_t)(m0 + n) * CD;
  v16h a;
#pragma unroll
  for (int c = 0; c < 8; ++c) {
    a[c]     = (_Float16)qrow[kh * 16 + c];
    a[8 + c] = (_Float16)qrow[kh * 16 + 8 + c];
  }

  float best[8];
  int   bidx[8];
#pragma unroll
  for (int j = 0; j < 8; ++j) { best[j] = -3.4e38f; bidx[j] = 0; }

  // stage chunk 0 into buffer 0
  {
    const char* gsrc = (const char*)embd_h;           // chunk 0 base
#pragma unroll
    for (int r = 0; r < 2; ++r) {
      int t = r * 256 + tid;                          // 0..511, 16B each
      async_cp16(eb0 + t * 16, gsrc + t * 16);
    }
    if (tid < 32) async_cp16(hb0 + tid * 16, (const char*)hn + tid * 16);
  }
  wait_async0();
  __syncthreads();

  for (int c = 0; c < NCHUNK; ++c) {
    const int cb = c & 1;
    // prefetch next chunk into the other buffer (async, overlaps WMMAs)
    if (c + 1 < NCHUNK) {
      const unsigned int edst = (cb ? eb0 : eb1);
      const unsigned int hdst = (cb ? hb0 : hb1);
      const char* gsrc = (const char*)(embd_h + (size_t)(c + 1) * CHUNK * CD);
      const char* hsrc = (const char*)(hn + (size_t)(c + 1) * CHUNK);
#pragma unroll
      for (int r = 0; r < 2; ++r) {
        int t = r * 256 + tid;
        async_cp16(edst + t * 16, gsrc + t * 16);
      }
      if (tid < 32) async_cp16(hdst + tid * 16, hsrc + tid * 16);
    }
    // consume current chunk: 8 N-tiles of 16 codes
    const _Float16* eb = &ebuf[cb][0];
    const float*    hb = &hbuf[cb][0];
#pragma unroll
    for (int tt = 0; tt < 8; ++tt) {
      const v16h b = *(const v16h*)(eb + ((tt * 16 + n) * CD + kh * 16));
      v8f acc = {};
      acc = __builtin_amdgcn_wmma_f32_16x16x32_f16(false, a, false, b, (short)0,
                                                   acc, false, false);
      const int   vc = c * CHUNK + tt * 16 + n;
      const float hv = hb[tt * 16 + n];
#pragma unroll
      for (int j = 0; j < 8; ++j) {
        float s  = acc[j] - hv;
        bool  up = s > best[j];
        best[j] = up ? s  : best[j];
        bidx[j] = up ? vc : bidx[j];
      }
    }
    wait_async0();
    __syncthreads();
  }

  // reduce across the 16 lanes of each half (each half shares its row set)
#pragma unroll
  for (int mask = 1; mask < 16; mask <<= 1) {
#pragma unroll
    for (int j = 0; j < 8; ++j) {
      float os = __shfl_xor(best[j], mask);
      int   oi = __shfl_xor(bidx[j], mask);
      bool take = (os > best[j]) || (os == best[j] && oi < bidx[j]);
      best[j] = take ? os : best[j];
      bidx[j] = take ? oi : bidx[j];
    }
  }
  if (n == 0) {
    const int mbase = m0 + kh * 8;   // lane0 -> rows 0..7, lane16 -> rows 8..15
#pragma unroll
    for (int j = 0; j < 8; ++j) idx_out[mbase + j] = bidx[j];
  }
}

// -------- K2: gather z_emb, decoder MLP -> z_recon, per-block loss partial ---
__global__ void k_dec(const int* __restrict__ idx, const float* __restrict__ embd,
                      const float* __restrict__ z2,
                      const float* __restrict__ w1, const float* __restrict__ b1,
                      const float* __restrict__ w2, const float* __restrict__ b2,
                      float* __restrict__ zrec, float* __restrict__ part) {
  __shared__ float w1s[CD * CH];   // 2048
  __shared__ float w2s[CH * CD];   // 2048
  __shared__ float red[256];
  const int tid = threadIdx.x;
  for (int i = tid; i < CD * CH; i += 256) { w1s[i] = w1[i]; w2s[i] = w2[i]; }
  __syncthreads();

  const int row = blockIdx.x * 256 + tid;          // 0..12799
  const int v   = idx[row];
  float ze[CD], h[CH];
  float lsum = 0.f;
#pragma unroll
  for (int d = 0; d < CD; ++d) {
    float e = embd[(size_t)v * CD + d];
    ze[d] = e;
    float df = z2[(size_t)row * CD + d] - e;
    lsum += df * df;
  }
#pragma unroll
  for (int j = 0; j < CH; ++j) {
    float s = b1[j];
#pragma unroll
    for (int d = 0; d < CD; ++d) s += ze[d] * w1s[d * CH + j];
    h[j] = s > 0.f ? s : 0.1f * s;
  }
#pragma unroll
  for (int d = 0; d < CD; ++d) {
    float s = b2[d];
#pragma unroll
    for (int j = 0; j < CH; ++j) s += h[j] * w2s[j * CD + d];
    zrec[(size_t)row * CD + d] = s;
  }
  red[tid] = lsum;
  __syncthreads();
  for (int off = 128; off > 0; off >>= 1) {
    if (tid < off) red[tid] += red[tid + off];
    __syncthreads();
  }
  if (tid == 0) part[blockIdx.x] = red[0];
}

// ---------------- K3: finalize loss ------------------------------------------
__global__ void k_loss(const float* __restrict__ part, float* __restrict__ out_loss) {
  if (threadIdx.x == 0) {
    float s = 0.f;
    for (int i = 0; i < CBS / 256; ++i) s += part[i];
    out_loss[0] = s / (float)(CBS * CD);
  }
}

// ---------------- K4: GRU, 31 steps, block = 64 batch rows -------------------
__global__ void k_gru(const float* __restrict__ x,    // (B,50,32)
                      const float* __restrict__ wih, const float* __restrict__ whh,
                      const float* __restrict__ bih, const float* __restrict__ bhh,
                      float* __restrict__ u_out) {
  __shared__ float h_s[64 * 32];
  __shared__ float gh_s[64 * 96];
  __shared__ float wih_s[96 * 32];
  __shared__ float whh_s[96 * 32];
  __shared__ float bih_s[96], bhh_s[96];
  const int tid = threadIdx.x;
  const int bb  = blockIdx.x * 64;
  for (int i = tid; i < 96 * 32; i += 256) { wih_s[i] = wih[i]; whh_s[i] = whh[i]; }
  for (int i = tid; i < 96; i += 256) { bih_s[i] = bih[i]; bhh_s[i] = bhh[i]; }
  for (int i = tid; i < 64 * 32; i += 256) h_s[i] = 0.f;
  __syncthreads();

  for (int t = 0; t <= CTRIG; ++t) {
    // pass1: gh = h @ whh^T + bhh  for 64x96
    for (int p = tid; p < 64 * 96; p += 256) {
      int bi = p / 96, e = p - bi * 96;
      gh_s[p] = bhh_s[e] + dot32(&h_s[bi * 32], &whh_s[e * 32]);
    }
    __syncthreads();
    // pass2: gates + state update for 64x32
    for (int p = tid; p < 64 * 32; p += 256) {
      int bi = p >> 5, d = p & 31;
      const float* xr = x + ((size_t)(bb + bi) * CS + t) * CD;
      float gir = bih_s[d]      + dot32(xr, &wih_s[d * 32]);
      float giz = bih_s[32 + d] + dot32(xr, &wih_s[(32 + d) * 32]);
      float gin = bih_s[64 + d] + dot32(xr, &wih_s[(64 + d) * 32]);
      float rr = sigm(gir + gh_s[bi * 96 + d]);
      float zz = sigm(giz + gh_s[bi * 96 + 32 + d]);
      float nn = tanhf(gin + rr * gh_s[bi * 96 + 64 + d]);
      h_s[p] = (1.f - zz) * nn + zz * h_s[p];
    }
    __syncthreads();
  }
  for (int p = tid; p < 64 * 32; p += 256) {
    int bi = p >> 5, d = p & 31;
    u_out[(size_t)(bb + bi) * CD + d] = h_s[p];
  }
}

// ---------------- K5: mapper MLPs (z1f / z2f) --------------------------------
__global__ void k_map(const float* __restrict__ u1, const float* __restrict__ u2,
                      const float* __restrict__ iw1, const float* __restrict__ ib1,
                      const float* __restrict__ iw2, const float* __restrict__ ib2,
                      const float* __restrict__ cw1, const float* __restrict__ cb1,
                      const float* __restrict__ cw2, const float* __restrict__ cb2,
                      float* __restrict__ z1f, float* __restrict__ z2f) {
  const int b = threadIdx.x;
  const float* u  = (blockIdx.x == 0) ? u1  : u2;
  const float* w1 = (blockIdx.x == 0) ? iw1 : cw1;
  const float* b1 = (blockIdx.x == 0) ? ib1 : cb1;
  const float* w2 = (blockIdx.x == 0) ? iw2 : cw2;
  const float* b2 = (blockIdx.x == 0) ? ib2 : cb2;
  float*       zf = (blockIdx.x == 0) ? z1f : z2f;

  float uv[CD], t1[CD];
#pragma unroll
  for (int d = 0; d < CD; ++d) uv[d] = u[b * CD + d];
#pragma unroll
  for (int j = 0; j < CD; ++j) {
    float s = b1[j];
#pragma unroll
    for (int d = 0; d < CD; ++d) s += uv[d] * w1[d * CD + j];
    t1[j] = tanhf(s);
  }
#pragma unroll
  for (int j = 0; j < CD; ++j) {
    float s = b2[j];
#pragma unroll
    for (int d = 0; d < CD; ++d) s += t1[d] * w2[d * CD + j];
    zf[b * CD + j] = s;
  }
}

// ---------------- K6: L=2 hypernetwork head ----------------------------------
__global__ void k_hyper(const float* __restrict__ x1,
                        const float* __restrict__ z1f, const float* __restrict__ z2f,
                        const float* __restrict__ hw1, const float* __restrict__ hb1,
                        const float* __restrict__ hw2, const float* __restrict__ hb2,
                        const float* __restrict__ lin_w, const float* __restrict__ lin_b,
                        const float* __restrict__ lout_w, const float* __restrict__ lout_b,
                        float* __restrict__ out) {
  const int b = threadIdx.x;
  float o[CU], zf1[CD], zf2[CD];
#pragma unroll
  for (int d = 0; d < CU; ++d) o[d] = x1[b * CU + d];
#pragma unroll
  for (int d = 0; d < CD; ++d) { zf1[d] = z1f[b * CD + d]; zf2[d] = z2f[b * CD + d]; }

  for (int i = 0; i < CL; ++i) {
    float y[CK];
#pragma unroll
    for (int k = 0; k < CK; ++k) {
      float s = lin_b[i * CK + k];
#pragma unroll
      for (int d = 0; d < CU; ++d) s += o[d] * lin_w[i * CU * CK + d * CK + k];
      y[k] = s;
    }
    float o2[CK];
#pragma unroll
    for (int j = 0; j < CK; ++j) {
      float acc = 0.f;
#pragma unroll
      for (int k = 0; k < CK; ++k) {
        int p = k * CK + j;
        float iw = hb1[i * CK * CK + p];
        float cw = hb2[i * CK * CK + p];
#pragma unroll
        for (int d = 0; d < CD; ++d) {
          iw += zf1[d] * hw1[i * CD * CK * CK + d * CK * CK + p];
          cw += zf2[d] * hw2[i * CD * CK * CK + d * CK * CK + p];
        }
        float w = fminf(fmaxf(iw, -CNORM), CNORM) + cw;
        acc += y[k] * w;
      }
      o2[j] = acc;
    }
#pragma unroll
    for (int d = 0; d < CU; ++d) {
      float s = lout_b[i * CU + d];
#pragma unroll
      for (int k = 0; k < CK; ++k) s += o2[k] * lout_w[i * CK * CU + k * CU + d];
      o[d] = s;
    }
  }
#pragma unroll
  for (int d = 0; d < CU; ++d) out[b * CU + d] = o[d];
}

// ---------------- launch ------------------------------------------------------
extern "C" void kernel_launch(void* const* d_in, const int* in_sizes, int n_in,
                              void* d_out, int out_size, void* d_ws, size_t ws_size,
                              hipStream_t stream) {
  const float* x1     = (const float*)d_in[0];
  const float* z1     = (const float*)d_in[1];
  const float* z2     = (const float*)d_in[3];
  const float* embd   = (const float*)d_in[4];
  const float* dec_w1 = (const float*)d_in[5];
  const float* dec_b1 = (const float*)d_in[6];
  const float* dec_w2 = (const float*)d_in[7];
  const float* dec_b2 = (const float*)d_in[8];
  const float* i_wih  = (const float*)d_in[9];
  const float* i_whh  = (const float*)d_in[10];
  const float* i_bih  = (const float*)d_in[11];
  const float* i_bhh  = (const float*)d_in[12];
  const float* c_wih  = (const float*)d_in[13];
  const float* c_whh  = (const float*)d_in[14];
  const float* c_bih  = (const float*)d_in[15];
  const float* c_bhh  = (const float*)d_in[16];
  const float* im_w1  = (const float*)d_in[17];
  const float* im_b1  = (const float*)d_in[18];
  const float* im_w2  = (const float*)d_in[19];
  const float* im_b2  = (const float*)d_in[20];
  const float* cm_w1  = (const float*)d_in[21];
  const float* cm_b1  = (const float*)d_in[22];
  const float* cm_w2  = (const float*)d_in[23];
  const float* cm_b2  = (const float*)d_in[24];
  const float* hw1    = (const float*)d_in[25];
  const float* hb1    = (const float*)d_in[26];
  const float* hw2    = (const float*)d_in[27];
  const float* hb2    = (const float*)d_in[28];
  const float* lin_w  = (const float*)d_in[29];
  const float* lin_b  = (const float*)d_in[30];
  const float* lout_w = (const float*)d_in[31];
  const float* lout_b = (const float*)d_in[32];

  char* ws = (char*)d_ws;
  _Float16* embd_h = (_Float16*)(ws + 0);            // 655360 B (padded V)
  float*    hn     = (float*)   (ws + 655360);       // 40960 B
  int*      idx    = (int*)     (ws + 696320);       // 51200 B
  float*    zrec   = (float*)   (ws + 747520);       // 1638400 B
  float*    part   = (float*)   (ws + 2385920);      // 200 B
  float*    u1     = (float*)   (ws + 2386176);      // 32768 B
  float*    u2     = (float*)   (ws + 2418944);      // 32768 B
  float*    z1f    = (float*)   (ws + 2451712);      // 32768 B
  float*    z2f    = (float*)   (ws + 2484480);      // 32768 B

  float* outp  = (float*)d_out;          // [0,8192) = out, [8192] = loss

  k_pack <<<CVP / 256, 256, 0, stream>>>(embd, embd_h, hn);
  k_vq   <<<MTILE / 8, 256, 0, stream>>>(z1, embd_h, hn, idx);
  k_dec  <<<CBS / 256, 256, 0, stream>>>(idx, embd, z2, dec_w1, dec_b1, dec_w2,
                                         dec_b2, zrec, part);
  k_loss <<<1, 64, 0, stream>>>(part, outp + CB * CU);
  k_gru  <<<CB / 64, 256, 0, stream>>>(zrec, i_wih, i_whh, i_bih, i_bhh, u1);
  k_gru  <<<CB / 64, 256, 0, stream>>>(z2,   c_wih, c_whh, c_bih, c_bhh, u2);
  k_map  <<<2, 256, 0, stream>>>(u1, u2, im_w1, im_b1, im_w2, im_b2,
                                 cm_w1, cm_b1, cm_w2, cm_b2, z1f, z2f);
  k_hyper<<<1, 256, 0, stream>>>(x1, z1f, z2f, hw1, hb1, hw2, hb2,
                                 lin_w, lin_b, lout_w, lout_b, outp);
}
